// Qwen3VLSequentialMoeTextExperts_22522808500778
// MI455X (gfx1250) — matrix operand order repaired and checked by reading, not compiled, not run on hardware
//
#include <hip/hip_runtime.h>
#include <math.h>

// Qwen3 MoE dense-over-experts (reference ignores router_indices).
// Two bf16-WMMA GEMM kernels for gfx1250, f32 accumulate, double-buffered LDS,
// async global->LDS staging via inline asm (portable across toolchains).

namespace {

constexpr int kE = 16;
constexpr int kH = 2048;   // hidden
constexpr int kI = 768;    // intermediate
constexpr int kT = 2048;   // B*S tokens

constexpr int TM = 64;     // token tile
constexpr int TN = 64;     // output-column tile
constexpr int TK = 32;     // K step (wmma 16x16x32 bf16)
constexpr int XSTR = 40;   // LDS row stride in halfwords (80B: 16B aligned, bank-spread)
constexpr int WSTR = 40;

typedef __bf16 bf16_t;
typedef bf16_t v16bf __attribute__((ext_vector_type(16)));
typedef float  v8f   __attribute__((ext_vector_type(8)));

#define USE_ASYNC_ASM 1

union Frag { v16bf v; unsigned short h[16]; uint2 d2[4]; uint4 d4[2]; };

// Pack two f32 -> two bf16 in one v_perm_b32 (+1 v_add each for rounding).
// 0x07060302 selects {S0[31:16], S1[31:16]} -> low half = a, high half = b.
__device__ __forceinline__ unsigned pack2bf(float a, float b) {
  union { float f; unsigned u; } x, y; x.f = a; y.f = b;
  return __builtin_amdgcn_perm(y.u + 0x8000u, x.u + 0x8000u, 0x07060302u);
}

__device__ __forceinline__ unsigned short f2bf(float f) {
  union { float f; unsigned u; } c; c.f = f;
  return (unsigned short)((c.u + 0x8000u) >> 16);   // round-to-nearest
}

__device__ __forceinline__ uint2 pack4(float a, float b, float c, float d) {
  uint2 r;
  r.x = pack2bf(a, b);
  r.y = pack2bf(c, d);
  return r;
}

__device__ __forceinline__ float silu_mul(float g, float u) {
  // silu(g)*u without IEEE divide: one v_exp + one v_rcp
  return g * __builtin_amdgcn_rcpf(1.0f + __expf(-g)) * u;
}

// Byte offset of a generic pointer to __shared__ within the wave's LDS space.
__device__ __forceinline__ unsigned lds_off32(void* p) {
  return (unsigned)(uintptr_t)(__attribute__((address_space(3))) void*)p;
}

// 16B async DMA global->LDS (GLOBAL_LOAD_ASYNC_TO_LDS_B128, tracked by ASYNCcnt).
__device__ __forceinline__ void async_copy16(const void* g, void* l) {
#if USE_ASYNC_ASM
  asm volatile("global_load_async_to_lds_b128 %0, %1, off"
               :
               : "v"(lds_off32(l)), "v"((unsigned long long)(uintptr_t)g)
               : "memory");
#else
  *reinterpret_cast<uint4*>(l) = *reinterpret_cast<const uint4*>(g);
#endif
}

__device__ __forceinline__ void wait_async0() {
#if USE_ASYNC_ASM
  asm volatile("s_wait_asynccnt 0x0" ::: "memory");
#endif
}

// A fragment (16x32 bf16): row=lane&15, K contiguous per lane-half (K = 16*half + j).
__device__ __forceinline__ void load_a(Frag& a, const unsigned short* lds_row, int half) {
  const unsigned short* p = lds_row + half * 16;
  a.d4[0] = *reinterpret_cast<const uint4*>(p + 0);
  a.d4[1] = *reinterpret_cast<const uint4*>(p + 8);
}

// B fragment (32x16 bf16): lane&15 = column n, K contiguous per lane-half.
__device__ __forceinline__ void load_b(Frag& b, const unsigned short* lds_row, int half) {
  const unsigned short* p = lds_row + half * 16;
  b.d4[0] = *reinterpret_cast<const uint4*>(p + 0);
  b.d4[1] = *reinterpret_cast<const uint4*>(p + 8);
}

// ---------------------------------------------------------------------------
// Kernel 1: gate_up GEMM + SwiGLU.  act[e,t,i] = silu(X@Wgᵀ) * (X@Wuᵀ), bf16.
// grid = (E, T/64, I/64), block = 256 (8 waves: 4 M x 2 N, 16x32 slab each,
// 4 wmma per K step). Double-buffered LDS: stage k+1 while computing k.
// ---------------------------------------------------------------------------
__global__ __launch_bounds__(256) void moe_gateup(
    const float* __restrict__ x,          // T x H
    const float* __restrict__ gw,         // E x 2I x H
    unsigned short* __restrict__ act) {   // E x T x I (bf16)
  __shared__ unsigned short xs[2][TM * XSTR];
  __shared__ unsigned short wg[2][TN * WSTR];
  __shared__ unsigned short wu[2][TN * WSTR];

  const int e     = blockIdx.x;
  const int tbase = blockIdx.y * TM;
  const int ibase = blockIdx.z * TN;

  const int tid  = threadIdx.x;
  const int lane = tid & 31;            // wave32
  const int wave = tid >> 5;
  const int wm   = wave & 3;            // M subtile
  const int wn   = wave >> 2;           // N half (32 cols)
  const int half = lane >> 4;
  const int l    = lane & 15;

  const int srow = tid >> 2;            // 0..63, 8 halfwords/thread
  const int scol = (tid & 3) * 8;

  auto stage = [&](int buf, int k0) {
    {   // X tile (f32 -> bf16)
      const float* p = x + (size_t)(tbase + srow) * kH + k0 + scol;
      float4 a4 = *(const float4*)p;
      float4 b4 = *(const float4*)(p + 4);
      uint4 q;
      q.x = pack2bf(a4.x, a4.y); q.y = pack2bf(a4.z, a4.w);
      q.z = pack2bf(b4.x, b4.y); q.w = pack2bf(b4.z, b4.w);
      *reinterpret_cast<uint4*>(&xs[buf][srow * XSTR + scol]) = q;
    }
    {   // gate + up weight tiles (64 n-rows x 32 k), f32 -> bf16
      const float* pg = gw + ((size_t)e * (2 * kI) + (ibase + srow)) * kH + k0 + scol;
      const float* pu = pg + (size_t)kI * kH;
      float4 g0 = *(const float4*)pg; float4 g1 = *(const float4*)(pg + 4);
      float4 u0 = *(const float4*)pu; float4 u1 = *(const float4*)(pu + 4);
      uint4 qg, qu;
      qg.x = pack2bf(g0.x, g0.y); qg.y = pack2bf(g0.z, g0.w);
      qg.z = pack2bf(g1.x, g1.y); qg.w = pack2bf(g1.z, g1.w);
      qu.x = pack2bf(u0.x, u0.y); qu.y = pack2bf(u0.z, u0.w);
      qu.z = pack2bf(u1.x, u1.y); qu.w = pack2bf(u1.z, u1.w);
      *reinterpret_cast<uint4*>(&wg[buf][srow * WSTR + scol]) = qg;
      *reinterpret_cast<uint4*>(&wu[buf][srow * WSTR + scol]) = qu;
    }
  };

  v8f cg0 = {}, cg1 = {}, cu0 = {}, cu1 = {};

  constexpr int NK = kH / TK;   // 64
  stage(0, 0);
  for (int i = 0; i < NK; ++i) {
    __syncthreads();
    if (i + 1 < NK) stage((i + 1) & 1, (i + 1) * TK);

    const int buf = i & 1;
    Frag a;
    load_a(a, &xs[buf][(wm * 16 + l) * XSTR], half);

    Frag b0, b1;
    load_b(b0, &wg[buf][(wn * 32 + l) * WSTR], half);
    load_b(b1, &wg[buf][(wn * 32 + 16 + l) * WSTR], half);
    cg0 = __builtin_amdgcn_wmma_f32_16x16x32_bf16(false, a.v, false, b0.v, (short)0, cg0, false, false);
    cg1 = __builtin_amdgcn_wmma_f32_16x16x32_bf16(false, a.v, false, b1.v, (short)0, cg1, false, false);

    load_b(b0, &wu[buf][(wn * 32 + l) * WSTR], half);
    load_b(b1, &wu[buf][(wn * 32 + 16 + l) * WSTR], half);
    cu0 = __builtin_amdgcn_wmma_f32_16x16x32_bf16(false, a.v, false, b0.v, (short)0, cu0, false, false);
    cu1 = __builtin_amdgcn_wmma_f32_16x16x32_bf16(false, a.v, false, b1.v, (short)0, cu1, false, false);
  }

  // SwiGLU epilogue; C layout: VGPR r -> M = r + 8*half, lane&15 -> N.
  const int trow = tbase + wm * 16 + half * 8;
  const int c0   = ibase + wn * 32 + l;
#pragma unroll
  for (int r = 0; r < 8; ++r) {
    const size_t base = ((size_t)e * kT + (trow + r)) * kI;
    act[base + c0]      = f2bf(silu_mul(cg0[r], cu0[r]));
    act[base + c0 + 16] = f2bf(silu_mul(cg1[r], cu1[r]));
  }
}

// ---------------------------------------------------------------------------
// Kernel 2: down GEMM + routed combine.
// out[t,h] = sum_e rw[t,e] * (act[e] @ Wd[e]ᵀ)[t,h]
// grid = (T/64, H/64), block = 256. Flattened (expert, kstep) loop,
// double-buffered LDS, act tiles staged via async global->LDS (bf16 copy).
// ---------------------------------------------------------------------------
__global__ __launch_bounds__(256) void moe_down(
    const unsigned short* __restrict__ act,  // E x T x I (bf16)
    const float* __restrict__ dw,            // E x H x I
    const float* __restrict__ rwt,           // T x E
    float* __restrict__ out) {               // T x H
  __shared__ unsigned short as_[2][TM * XSTR];
  __shared__ unsigned short ws_[2][TN * WSTR];

  const int tbase = blockIdx.x * TM;
  const int hbase = blockIdx.y * TN;

  const int tid  = threadIdx.x;
  const int lane = tid & 31;
  const int wave = tid >> 5;
  const int wm   = wave & 3;
  const int wn   = wave >> 2;
  const int half = lane >> 4;
  const int l    = lane & 15;

  const int srow = tid >> 2;
  const int scol = (tid & 3) * 8;

  constexpr int KSTEPS = kI / TK;           // 24
  constexpr int NSTEP  = kE * KSTEPS;       // 384

  auto stage = [&](int buf, int step) {
    const int e  = step / KSTEPS;
    const int k0 = (step % KSTEPS) * TK;
    // act tile: straight bf16 16B copy -> async DMA to LDS
    async_copy16(&act[((size_t)e * kT + tbase + srow) * kI + k0 + scol],
                 &as_[buf][srow * XSTR + scol]);
    // down-weight tile: B[k=i][n=h] from dw[e, hbase+n, k0+i], f32 -> bf16
    const float* pw = dw + ((size_t)e * kH + hbase + srow) * kI + k0 + scol;
    float4 w0 = *(const float4*)pw; float4 w1 = *(const float4*)(pw + 4);
    uint4 q;
    q.x = pack2bf(w0.x, w0.y); q.y = pack2bf(w0.z, w0.w);
    q.z = pack2bf(w1.x, w1.y); q.w = pack2bf(w1.z, w1.w);
    *reinterpret_cast<uint4*>(&ws_[buf][srow * WSTR + scol]) = q;
  };

  v8f acc0 = {}, acc1 = {}, ce0 = {}, ce1 = {};
  const int trow = tbase + wm * 16 + half * 8;

  stage(0, 0);
  for (int step = 0; step < NSTEP; ++step) {
    wait_async0();     // our async tiles landed in LDS (ASYNCcnt -> 0)
    __syncthreads();   // ...and everyone else's
    if (step + 1 < NSTEP) stage((step + 1) & 1, step + 1);

    const int buf = step & 1;
    Frag a;
    load_a(a, &as_[buf][(wm * 16 + l) * XSTR], half);
    Frag b0, b1;
    load_b(b0, &ws_[buf][(wn * 32 + l) * WSTR], half);
    load_b(b1, &ws_[buf][(wn * 32 + 16 + l) * WSTR], half);
    ce0 = __builtin_amdgcn_wmma_f32_16x16x32_bf16(false, a.v, false, b0.v, (short)0, ce0, false, false);
    ce1 = __builtin_amdgcn_wmma_f32_16x16x32_bf16(false, a.v, false, b1.v, (short)0, ce1, false, false);

    if ((step % KSTEPS) == KSTEPS - 1) {
      // end of expert: routed combine, reset per-expert accumulators
      const int e = step / KSTEPS;
#pragma unroll
      for (int r = 0; r < 8; ++r) {
        const float rw = rwt[(size_t)(trow + r) * kE + e];
        acc0[r] += rw * ce0[r];
        acc1[r] += rw * ce1[r];
      }
      ce0 = {}; ce1 = {};
    }
  }

  const int c0 = hbase + wn * 32 + l;
#pragma unroll
  for (int r = 0; r < 8; ++r) {
    out[(size_t)(trow + r) * kH + c0]      = acc0[r];
    out[(size_t)(trow + r) * kH + c0 + 16] = acc1[r];
  }
}

}  // namespace

extern "C" void kernel_launch(void* const* d_in, const int* in_sizes, int n_in,
                              void* d_out, int out_size, void* d_ws, size_t ws_size,
                              hipStream_t stream) {
  (void)in_sizes; (void)n_in; (void)out_size; (void)ws_size;
  const float* x   = (const float*)d_in[0];   // hidden_states (B,S,H) -> T x H
  const float* rwt = (const float*)d_in[1];   // routing_weights T x E
  // d_in[2] = router_indices: unused by the reference computation
  const float* gw  = (const float*)d_in[3];   // gate_up_w E x 2I x H
  const float* dw  = (const float*)d_in[4];   // down_w    E x H x I

  float* out = (float*)d_out;                  // T x H f32
  unsigned short* actbuf = (unsigned short*)d_ws; // E*T*I bf16 = 48 MiB scratch

  dim3 g1(kE, kT / TM, kI / TN);   // (16, 32, 12)
  moe_gateup<<<g1, dim3(256), 0, stream>>>(x, gw, actbuf);

  dim3 g2(kT / TM, kH / TN);       // (32, 32)
  moe_down<<<g2, dim3(256), 0, stream>>>(actbuf, dw, rwt, out);
}
